// FragmentGNN_56813827392049
// MI455X (gfx1250) — matrix operand order repaired and checked by reading, not compiled, not run on hardware
//
#include <hip/hip_runtime.h>

#define NNODE 50000
#define NEDGE 500000

typedef __attribute__((ext_vector_type(16))) __bf16 v16bf;
typedef __attribute__((ext_vector_type(8)))  float  v8f;

union ABu { unsigned int u[8]; v16bf v; };

__device__ __forceinline__ unsigned short bf16_bits(float a){
  union { __bf16 h; unsigned short s; } p;
  p.h = (__bf16)a;
  return p.s;
}
__device__ __forceinline__ unsigned int pack2_bf16(float a, float b){
  union { __bf16 h[2]; unsigned int u; } p;
  p.h[0] = (__bf16)a; p.h[1] = (__bf16)b;
  return p.u;
}

__device__ __forceinline__ float fast_rcp(float x){ return __builtin_amdgcn_rcpf(x); }
__device__ __forceinline__ float fast_rsq(float x){ return __builtin_amdgcn_rsqf(x); }
__device__ __forceinline__ float silu_f(float x){ return x * fast_rcp(1.f + __expf(-x)); }
__device__ __forceinline__ float sigmoid_f(float x){ return fast_rcp(1.f + __expf(-x)); }

__device__ __forceinline__ v8f zero8(){
  v8f c;
  #pragma unroll
  for(int i=0;i<8;i++) c[i]=0.f;
  return c;
}

__device__ __forceinline__ v8f wmma_b(const ABu& a, const ABu& b, v8f c){
  return __builtin_amdgcn_wmma_f32_16x16x32_bf16(false, a.v, false, b.v, (short)0, c,
                                                 false, false);
}

// A frag (bf16 LDS, row stride SA ushorts, 16B-multiple): lane m=lane&15, kh=lane>>4;
//   VGPR v holds K = kt*32 + kh*16 + 2v (+1)  -> dwords {0..7} from base
// B frag (global, [lane][v] contiguous): VGPR v holds K = kt*32 + kh*16 + 2v (+1), col n0+(lane&15)
__device__ __forceinline__ void load_b(ABu& b, const unsigned int* __restrict__ Wf,
                                       int kt, int nt, int lane){
  const uint4* bp = (const uint4*)(Wf + (size_t)(kt*8 + nt)*256 + lane*8);
  *(uint4*)&b.u[0] = bp[0];
  *(uint4*)&b.u[4] = bp[1];
}
__device__ __forceinline__ void load_a(ABu& a, const unsigned short* As, int row, int SA,
                                       int kt, int kh){
  const uint4* ap = (const uint4*)(As + row*SA + kt*32 + kh*16);
  *(uint4*)&a.u[0] = ap[0];
  *(uint4*)&a.u[4] = ap[1];
}

// one 16x16 tile per wave (M=16)
__device__ __forceinline__ v8f gemm_tile(const unsigned short* As, int SA,
                                         const unsigned int* __restrict__ Wf,
                                         int KT, int nt, int lane){
  v8f c = zero8();
  int m = lane & 15, kh = lane >> 4;
  for(int kt = 0; kt < KT; ++kt){
    ABu a, b;
    load_b(b, Wf, kt, nt, lane);
    load_a(a, As, m, SA, kt, kh);
    c = wmma_b(a, b, c);
  }
  return c;
}

// two 16x16 tiles per wave (M=32), B reused, independent accumulators
__device__ __forceinline__ void gemm_tile2(const unsigned short* As, int SA,
                                           const unsigned int* __restrict__ Wf,
                                           int KT, int nt, int lane, v8f& cA, v8f& cB){
  cA = zero8(); cB = zero8();
  int m = lane & 15, kh = lane >> 4;
  for(int kt = 0; kt < KT; ++kt){
    ABu a0, a1, b;
    load_b(b, Wf, kt, nt, lane);
    load_a(a0, As, m,      SA, kt, kh);
    load_a(a1, As, m + 16, SA, kt, kh);
    cA = wmma_b(a0, b, cA);
    cB = wmma_b(a1, b, cB);
  }
}

// C layout: VGPR i -> row rbase + i + 8*(lane>>4), col n0 + (lane&15)
__device__ __forceinline__ void store_hidden_silu(v8f c, const float* __restrict__ bias,
                                                  unsigned short* Hs, int SH, int n0,
                                                  int lane, int rbase){
  int n = n0 + (lane & 15);
  float bv = bias[n];
  int roff = rbase + (lane >> 4) * 8;
  #pragma unroll
  for(int i=0;i<8;i++)
    Hs[(roff+i)*SH + n] = bf16_bits(silu_f(c[i] + bv));
}

__device__ __forceinline__ void store_out_silu(v8f c, const float* __restrict__ bias,
                                               float* Os, int n0, int lane, int rbase){
  int n = n0 + (lane & 15);
  float bv = bias[n];
  int roff = rbase + (lane >> 4) * 8;
  #pragma unroll
  for(int i=0;i<8;i++)
    Os[(roff+i)*128 + n] = silu_f(c[i] + bv);
}

// gather [h_src|h_dst|h_edge] as float4, convert+pack to bf16, 8B LDS stores
template<int ROWS>
__device__ __forceinline__ void gather384(unsigned short* As,
    const float* __restrict__ h_node, const float* __restrict__ h_edge,
    const int* ssrc, const int* sdst, int e0, int tid){
  for(int i = tid; i < ROWS*96; i += 256){
    int r = i / 96, q = i - r*96;
    const float* base;
    if(q < 32)      base = h_node + (size_t)ssrc[r]*128 + q*4;
    else if(q < 64) base = h_node + (size_t)sdst[r]*128 + (q-32)*4;
    else            base = h_edge + (size_t)(e0+r)*128 + (q-64)*4;
    float4 f = *(const float4*)base;
    uint2 pk;
    pk.x = pack2_bf16(f.x, f.y);
    pk.y = pack2_bf16(f.z, f.w);
    *(uint2*)(As + r*392 + q*4) = pk;
  }
}

// ROWS x 128 LayerNorm; 16 threads per row, shfl within 16-lane groups (wave32 halves)
template<int ROWS>
__device__ __forceinline__ void layernorm_write(const float* Os,
    const float* __restrict__ g, const float* __restrict__ b,
    float* __restrict__ out, int tid){
  #pragma unroll
  for(int half=0; half<ROWS/16; ++half){
    int r = (tid >> 4) + half*16;
    int c0 = (tid & 15) * 8;
    float x[8], s = 0.f, s2 = 0.f;
    #pragma unroll
    for(int i=0;i<8;i++){ x[i] = Os[r*128 + c0 + i]; s += x[i]; s2 += x[i]*x[i]; }
    #pragma unroll
    for(int off=1; off<16; off<<=1){ s += __shfl_xor(s, off, 16); s2 += __shfl_xor(s2, off, 16); }
    float mu  = s  * (1.f/128.f);
    float var = s2 * (1.f/128.f) - mu*mu;
    float rs  = fast_rsq(var + 1e-5f);
    #pragma unroll
    for(int i=0;i<8;i++){
      int c = c0 + i;
      out[(size_t)r*128 + c] = (x[i]-mu)*rs*g[c] + b[c];
    }
  }
}

template<int ROWS>
__device__ __forceinline__ void head_dot(const float* Os, const float* __restrict__ w2,
    const float* __restrict__ b2, float* __restrict__ out, int e0, int tid, bool sig){
  #pragma unroll
  for(int half=0; half<ROWS/16; ++half){
    int r = (tid >> 4) + half*16;
    int c0 = (tid & 15) * 8;
    float s = 0.f;
    #pragma unroll
    for(int i=0;i<8;i++) s += Os[r*128 + c0 + i] * w2[c0 + i];
    #pragma unroll
    for(int off=1; off<16; off<<=1) s += __shfl_xor(s, off, 16);
    if((tid & 15) == 0){
      float v = s + b2[0];
      if(sig) v = sigmoid_f(v);
      out[e0 + r] = v;
    }
  }
}

// ---- Convert f32 weight [Kdin x 128] to bf16 fragments, [lane][v] contiguous per tile ----
__global__ void __launch_bounds__(256) k_prep(const float* __restrict__ W, int Kdin, int total,
                                              unsigned int* __restrict__ out){
  int idx = blockIdx.x*256 + threadIdx.x;
  if(idx >= total) return;
  int v    =  idx & 7;
  int lane = (idx >> 3) & 31;
  int nt   = (idx >> 8) & 7;
  int kt   =  idx >> 11;
  int n  = nt*16 + (lane & 15);
  int kh = lane >> 4;
  int k1 = kt*32 + kh*16 + v*2;
  float lo = (k1   < Kdin) ? W[(size_t)k1*128 + n]     : 0.f;
  float hi = (k1+1 < Kdin) ? W[(size_t)(k1+1)*128 + n] : 0.f;
  out[idx] = pack2_bf16(lo, hi);
}

// ---- Embedding MLP2: X [M x Fin] -> out [M x 128], Fin <= 32, M=16/block ----
__global__ void __launch_bounds__(256) k_embed(const float* __restrict__ X, int Fin,
    const unsigned int* __restrict__ Wf0, const float* __restrict__ b0,
    const unsigned int* __restrict__ Wf1, const float* __restrict__ b1,
    float* __restrict__ out){
  __shared__ __align__(16) unsigned short As[16*40];
  __shared__ __align__(16) unsigned short Hs[16*136];
  int tid = threadIdx.x;
  int r0 = blockIdx.x * 16;
  for(int i=tid; i<512; i+=256){
    int r = i >> 5, c = i & 31;
    float v = (c < Fin) ? X[(size_t)(r0+r)*Fin + c] : 0.f;
    As[r*40 + c] = bf16_bits(v);
  }
  __syncthreads();
  int lane = tid & 31, w = tid >> 5;
  v8f c1 = gemm_tile(As, 40, Wf0, 1, w, lane);
  store_hidden_silu(c1, b0, Hs, 136, w*16, lane, 0);
  __syncthreads();
  v8f c2 = gemm_tile(Hs, 136, Wf1, 4, w, lane);
  int n = w*16 + (lane & 15);
  float bv = b1[n];
  int roff = (lane >> 4) * 8;
  #pragma unroll
  for(int i=0;i<8;i++)
    out[(size_t)(r0 + roff + i)*128 + n] = c2[i] + bv;
}

// ---- msg MLP2 over edges (M=32/block) + atomic scatter into agg ----
__global__ void __launch_bounds__(256) k_msg(
    const float* __restrict__ h_node, const float* __restrict__ h_edge,
    const int* __restrict__ eidx,
    const unsigned int* __restrict__ Wf0, const float* __restrict__ b0,
    const unsigned int* __restrict__ Wf1, const float* __restrict__ b1,
    float* __restrict__ agg){
  __shared__ __align__(16) unsigned short As[32*392];
  __shared__ __align__(16) unsigned short Hs[32*136];
  __shared__ int ssrc[32], sdst[32];
  int tid = threadIdx.x;
  int e0 = blockIdx.x * 32;
  if(tid < 32){ ssrc[tid]=eidx[(size_t)(e0+tid)*2]; sdst[tid]=eidx[(size_t)(e0+tid)*2+1]; }
  __syncthreads();
  gather384<32>(As, h_node, h_edge, ssrc, sdst, e0, tid);
  __syncthreads();
  int lane = tid & 31, w = tid >> 5;
  v8f cA, cB;
  gemm_tile2(As, 392, Wf0, 12, w, lane, cA, cB);
  store_hidden_silu(cA, b0, Hs, 136, w*16, lane, 0);
  store_hidden_silu(cB, b0, Hs, 136, w*16, lane, 16);
  __syncthreads();
  gemm_tile2(Hs, 136, Wf1, 4, w, lane, cA, cB);
  int n = w*16 + (lane & 15);
  float bv = b1[n];
  int roff = (lane >> 4) * 8;
  #pragma unroll
  for(int i=0;i<8;i++){
    unsafeAtomicAdd(&agg[(size_t)sdst[roff+i]*128 + n],    cA[i] + bv);
    unsafeAtomicAdd(&agg[(size_t)sdst[16+roff+i]*128 + n], cB[i] + bv);
  }
}

// ---- node update: LN(h + MLP2([h|agg])), M=16/block ----
__global__ void __launch_bounds__(256) k_node(
    float* __restrict__ h_node, const float* __restrict__ agg,
    const unsigned int* __restrict__ Wf0, const float* __restrict__ b0,
    const unsigned int* __restrict__ Wf1, const float* __restrict__ b1,
    const float* __restrict__ g, const float* __restrict__ bln){
  __shared__ __align__(16) unsigned short As[16*264];
  __shared__ __align__(16) unsigned short Hs[16*136];
  __shared__ __align__(16) float Os[16*128];
  int tid = threadIdx.x;
  int r0 = blockIdx.x * 16;
  for(int i = tid; i < 16*64; i += 256){          // 64 float4 per row (K=256)
    int r = i >> 6, q = i & 63;
    const float* base = (q < 32) ? (h_node + (size_t)(r0+r)*128 + q*4)
                                 : (agg    + (size_t)(r0+r)*128 + (q-32)*4);
    float4 f = *(const float4*)base;
    uint2 pk;
    pk.x = pack2_bf16(f.x, f.y);
    pk.y = pack2_bf16(f.z, f.w);
    *(uint2*)(As + r*264 + q*4) = pk;
  }
  __syncthreads();
  int lane = tid & 31, w = tid >> 5;
  v8f c1 = gemm_tile(As, 264, Wf0, 8, w, lane);
  store_hidden_silu(c1, b0, Hs, 136, w*16, lane, 0);
  __syncthreads();
  v8f c2 = gemm_tile(Hs, 136, Wf1, 4, w, lane);
  int n = w*16 + (lane & 15);
  float bv = b1[n];
  int roff = (lane >> 4) * 8;
  #pragma unroll
  for(int i=0;i<8;i++)
    Os[(roff+i)*128 + n] = h_node[(size_t)(r0+roff+i)*128 + n] + c2[i] + bv;
  __syncthreads();
  layernorm_write<16>(Os, g, bln, h_node + (size_t)r0*128, tid);
}

// ---- edge update: LN(h_e + MLP2([h_s|h_d|h_e])), M=32/block ----
__global__ void __launch_bounds__(256) k_edge(
    const float* __restrict__ h_node, float* __restrict__ h_edge,
    const int* __restrict__ eidx,
    const unsigned int* __restrict__ Wf0, const float* __restrict__ b0,
    const unsigned int* __restrict__ Wf1, const float* __restrict__ b1,
    const float* __restrict__ g, const float* __restrict__ bln){
  __shared__ __align__(16) unsigned short As[32*392];
  __shared__ __align__(16) unsigned short Hs[32*136];
  __shared__ __align__(16) float Os[32*128];
  __shared__ int ssrc[32], sdst[32];
  int tid = threadIdx.x;
  int e0 = blockIdx.x * 32;
  if(tid < 32){ ssrc[tid]=eidx[(size_t)(e0+tid)*2]; sdst[tid]=eidx[(size_t)(e0+tid)*2+1]; }
  __syncthreads();
  gather384<32>(As, h_node, h_edge, ssrc, sdst, e0, tid);
  __syncthreads();
  int lane = tid & 31, w = tid >> 5;
  v8f cA, cB;
  gemm_tile2(As, 392, Wf0, 12, w, lane, cA, cB);
  store_hidden_silu(cA, b0, Hs, 136, w*16, lane, 0);
  store_hidden_silu(cB, b0, Hs, 136, w*16, lane, 16);
  __syncthreads();
  gemm_tile2(Hs, 136, Wf1, 4, w, lane, cA, cB);
  int n = w*16 + (lane & 15);
  float bv = b1[n];
  int roff = (lane >> 4) * 8;
  #pragma unroll
  for(int i=0;i<8;i++){
    Os[(roff+i)*128 + n]    = h_edge[(size_t)(e0+roff+i)*128 + n]    + cA[i] + bv;
    Os[(16+roff+i)*128 + n] = h_edge[(size_t)(e0+16+roff+i)*128 + n] + cB[i] + bv;
  }
  __syncthreads();
  layernorm_write<32>(Os, g, bln, h_edge + (size_t)e0*128, tid);
}

// ---- both MLP3 heads, shared gather, M=32/block ----
__global__ void __launch_bounds__(256) k_heads(
    const float* __restrict__ h_node, const float* __restrict__ h_edge,
    const int* __restrict__ eidx,
    const unsigned int* __restrict__ mWf0, const float* __restrict__ mb0,
    const unsigned int* __restrict__ mWf1, const float* __restrict__ mb1,
    const float* __restrict__ mw2, const float* __restrict__ mb2,
    const unsigned int* __restrict__ rWf0, const float* __restrict__ rb0,
    const unsigned int* __restrict__ rWf1, const float* __restrict__ rb1,
    const float* __restrict__ rw2, const float* __restrict__ rb2,
    float* __restrict__ out_m, float* __restrict__ out_r){
  __shared__ __align__(16) unsigned short As[32*392];
  __shared__ __align__(16) unsigned short Hs[32*136];
  __shared__ __align__(16) float Os[32*128];
  __shared__ int ssrc[32], sdst[32];
  int tid = threadIdx.x;
  int e0 = blockIdx.x * 32;
  if(tid < 32){ ssrc[tid]=eidx[(size_t)(e0+tid)*2]; sdst[tid]=eidx[(size_t)(e0+tid)*2+1]; }
  __syncthreads();
  gather384<32>(As, h_node, h_edge, ssrc, sdst, e0, tid);
  __syncthreads();
  int lane = tid & 31, w = tid >> 5;
  v8f cA, cB;
  { // merge head
    gemm_tile2(As, 392, mWf0, 12, w, lane, cA, cB);
    store_hidden_silu(cA, mb0, Hs, 136, w*16, lane, 0);
    store_hidden_silu(cB, mb0, Hs, 136, w*16, lane, 16);
    __syncthreads();
    gemm_tile2(Hs, 136, mWf1, 4, w, lane, cA, cB);
    store_out_silu(cA, mb1, Os, w*16, lane, 0);
    store_out_silu(cB, mb1, Os, w*16, lane, 16);
    __syncthreads();
    head_dot<32>(Os, mw2, mb2, out_m, e0, tid, false);
    __syncthreads();
  }
  { // risk head
    gemm_tile2(As, 392, rWf0, 12, w, lane, cA, cB);
    store_hidden_silu(cA, rb0, Hs, 136, w*16, lane, 0);
    store_hidden_silu(cB, rb0, Hs, 136, w*16, lane, 16);
    __syncthreads();
    gemm_tile2(Hs, 136, rWf1, 4, w, lane, cA, cB);
    store_out_silu(cA, rb1, Os, w*16, lane, 0);
    store_out_silu(cB, rb1, Os, w*16, lane, 16);
    __syncthreads();
    head_dot<32>(Os, rw2, rb2, out_r, e0, tid, true);
  }
}

extern "C" void kernel_launch(void* const* d_in, const int* in_sizes, int n_in,
                              void* d_out, int out_size, void* d_ws, size_t ws_size,
                              hipStream_t stream){
  (void)in_sizes; (void)n_in; (void)out_size; (void)ws_size;
  const float* node_feat = (const float*)d_in[0];
  const int*   eidx      = (const int*)d_in[1];
  const float* edge_feat = (const float*)d_in[2];
  auto F = [&](int i){ return (const float*)d_in[i]; };

  // workspace layout
  float* h_node = (float*)d_ws;                         // 50000*128 f32
  float* agg    = h_node + (size_t)NNODE*128;           // 50000*128 f32
  float* h_edge = agg    + (size_t)NNODE*128;           // 500000*128 f32
  unsigned int* wf = (unsigned int*)(h_edge + (size_t)NEDGE*128); // bf16 weight frags

  size_t cur = 0;
  auto prep = [&](int widx, int Kdin)->const unsigned int*{
    int KT = (Kdin + 31) / 32;
    int total = KT * 2048;                 // KT * 8nt * 32lane * 8v dwords
    unsigned int* p = wf + cur;
    cur += (size_t)total;
    k_prep<<<(total+255)/256, 256, 0, stream>>>(F(widx), Kdin, total, p);
    return p;
  };

  // params flattened in insertion order: node_embed, edge_embed, layers[6], merge_head, risk_head
  // lin = {w,b}; layer = {msg, upd, edge_upd, node_norm{g,b}, edge_norm{g,b}}
  const unsigned int* ne_w0 = prep(3, 12);
  const unsigned int* ne_w1 = prep(5, 128);
  const unsigned int* ee_w0 = prep(7, 3);
  const unsigned int* ee_w1 = prep(9, 128);
  const unsigned int *msg0[6], *msg1[6], *upd0[6], *upd1[6], *eu0[6], *eu1[6];
  for(int l=0;l<6;l++){
    int lb = 11 + 16*l;
    msg0[l] = prep(lb+0, 384);
    msg1[l] = prep(lb+2, 128);
    upd0[l] = prep(lb+4, 256);
    upd1[l] = prep(lb+6, 128);
    eu0[l]  = prep(lb+8, 384);
    eu1[l]  = prep(lb+10, 128);
  }
  const unsigned int* m_w0 = prep(107, 384);
  const unsigned int* m_w1 = prep(109, 128);
  const unsigned int* r_w0 = prep(113, 384);
  const unsigned int* r_w1 = prep(115, 128);

  // embeddings
  k_embed<<<NNODE/16, 256, 0, stream>>>(node_feat, 12, ne_w0, F(4),  ne_w1, F(6),  h_node);
  k_embed<<<NEDGE/16, 256, 0, stream>>>(edge_feat, 3,  ee_w0, F(8),  ee_w1, F(10), h_edge);

  // message-passing layers
  for(int l=0;l<6;l++){
    int lb = 11 + 16*l;
    hipMemsetAsync(agg, 0, (size_t)NNODE*128*sizeof(float), stream);
    k_msg <<<NEDGE/32, 256, 0, stream>>>(h_node, h_edge, eidx,
        msg0[l], F(lb+1), msg1[l], F(lb+3), agg);
    k_node<<<NNODE/16, 256, 0, stream>>>(h_node, agg,
        upd0[l], F(lb+5), upd1[l], F(lb+7), F(lb+12), F(lb+13));
    k_edge<<<NEDGE/32, 256, 0, stream>>>(h_node, h_edge, eidx,
        eu0[l], F(lb+9), eu1[l], F(lb+11), F(lb+14), F(lb+15));
  }

  // heads
  k_heads<<<NEDGE/32, 256, 0, stream>>>(h_node, h_edge, eidx,
      m_w0, F(108), m_w1, F(110), F(111), F(112),
      r_w0, F(114), r_w1, F(116), F(117), F(118),
      (float*)d_out, (float*)d_out + NEDGE);
}
